// PairwiseInteractions_55087250539205
// MI455X (gfx1250) — compile-verified, matrix operand order checked by reading, not run on hardware
//
#include <hip/hip_runtime.h>
#include <math.h>
#include <stdint.h>

// ---------------------------------------------------------------------------
// Problem constants (from the reference)
// ---------------------------------------------------------------------------
#define BATCH     2048
#define NEGS      512
#define INPUT_DIM 256
#define DIM       64
#define VOCAB     32
#define N_HEADS   5
#define N_PAIR    6

// LDS padding strides (floats): stride mod 64 == 4 -> 16 consecutive rows hit
// 16 distinct banks; stride mod 4 == 0 keeps float4 (b128) accesses aligned.
// TDM reproduces these with pad_enable (pad 4 dwords per row).
#define XS_STR   260   // x tile rows in kernel 1   (256 + 4 pad dwords)
#define EMB_STR  68    // emb rows in kernel 2      (64 + 4 pad dwords)

typedef __attribute__((ext_vector_type(16))) __bf16        v16bf;
typedef __attribute__((ext_vector_type(8)))  float         v8f;
typedef __attribute__((ext_vector_type(4)))  unsigned int  v4u;
typedef __attribute__((ext_vector_type(8)))  int           v8i;
typedef __attribute__((ext_vector_type(4)))  int           v4i;

// gfx1250 has a hardware v_tanh_f32 (TRANS op); use it when exposed.
#if __has_builtin(__builtin_amdgcn_tanhf)
#define FAST_TANH(x) __builtin_amdgcn_tanhf(x)
#else
#define FAST_TANH(x) tanhf(x)
#endif

// Tensor Data Mover (gfx1250). This toolchain exposes the 6-arg form:
//   (uint32x4 g0, int32x8 g1, int32x4 g2, int32x4 g3, int32x8 g4, i32 cpol)
#if __has_builtin(__builtin_amdgcn_tensor_load_to_lds) && \
    __has_builtin(__builtin_amdgcn_s_wait_tensorcnt)
#define HAVE_TDM 1
#else
#define HAVE_TDM 0
#endif

// Low 32 bits of a flat pointer into the shared aperture are the LDS byte
// address (ISA 10.2).
__device__ __forceinline__ unsigned lds_byte_addr(const void* p) {
    return (unsigned)(uintptr_t)p;
}

#if HAVE_TDM
// Issue one TDM tile load described per CDNA5 ISA.
// 2-D when tile2==0, 3-D otherwise. data_size fixed at 4 bytes (code 2).
// Address calc: global = base + 4*(x + y*stride0 + z*stride1).
__device__ __forceinline__ void tdm_load_tile(
    unsigned lds_addr, const void* gptr,
    unsigned dim0, unsigned dim1, unsigned dim2,
    unsigned tile0, unsigned tile1, unsigned tile2,
    unsigned long long stride0, unsigned long long stride1,
    int pad_interval, int pad_amount, int pad_en)
{
    const unsigned long long ga = (unsigned long long)gptr;
    // Group 0: count=1 | lds_addr | global_addr[56:0] | type=2 (bits 127:126)
    const v4u g0 = { 1u,
                     lds_addr,
                     (unsigned)(ga & 0xffffffffu),
                     (unsigned)((ga >> 32) & 0x01ffffffu) | (2u << 30) };
    // Group 1
    const int d0 = (2 << 16)                 // data_size = 4 bytes
                 | (pad_en << 20)
                 | (pad_interval << 22)
                 | (pad_amount << 25);
    const v8i g1 = {
        d0,
        (int)((dim0 & 0xffffu) << 16),                                   // [63:48] dim0 lo
        (int)(((dim0 >> 16) & 0xffffu) | ((dim1 & 0xffffu) << 16)),      // dim0 hi | dim1 lo
        (int)(((dim1 >> 16) & 0xffffu) | ((tile0 & 0xffffu) << 16)),     // dim1 hi | tile0
        (int)((tile1 & 0xffffu) | ((tile2 & 0xffffu) << 16)),            // tile1 | tile2
        (int)(stride0 & 0xffffffffu),                                    // stride0 lo
        (int)(((stride0 >> 32) & 0xffffu) | ((stride1 & 0xffffu) << 16)),// s0 hi | s1 lo
        (int)((stride1 >> 16) & 0xffffffffu) };                          // stride1 hi
    // Group 2: tensor_dim2 (length of z); rest unused. Groups 3/4: zero.
    const v4i g2 = { (int)dim2, 0, 0, 0 };
    const v4i g3 = { 0, 0, 0, 0 };
    const v8i g4 = { 0, 0, 0, 0, 0, 0, 0, 0 };
    __builtin_amdgcn_tensor_load_to_lds(g0, g1, g2, g3, g4, 0);
}
#endif

// Pack 4 f32 into 4 consecutive bf16 elements of a fragment.
__device__ __forceinline__ void put4(v16bf& v, int base, float4 f) {
    v[base + 0] = (__bf16)f.x;
    v[base + 1] = (__bf16)f.y;
    v[base + 2] = (__bf16)f.z;
    v[base + 3] = (__bf16)f.w;
}
__device__ __forceinline__ float4 mul4(float4 a, float4 b) {
    return make_float4(a.x * b.x, a.y * b.y, a.z * b.z, a.w * b.w);
}

// CDNA5 16-bit A/B fragment layout: for lane-half hf, element e maps to
//   e = 0..7  -> K = kb + e          (kb = kstep + hf*8)
//   e = 8..15 -> K = kb + 16 + (e-8)
// i.e. two aligned 8-float runs -> 4 float4 loads per fragment.
__device__ __forceinline__ void frag_from(v16bf& v, const float* row, int kb) {
    put4(v, 0,  *(const float4*)&row[kb]);
    put4(v, 4,  *(const float4*)&row[kb + 4]);
    put4(v, 8,  *(const float4*)&row[kb + 16]);
    put4(v, 12, *(const float4*)&row[kb + 20]);
}

// ---------------------------------------------------------------------------
// Kernel 1: gates  g[p,b,d] = tanh( sum_i x[b,i]*gate_w[p,d,i] + gate_b[p,d] )
// grid (BATCH/16, N_PAIR), block 128 (4 waves); each wave owns a 16-wide d tile.
// ---------------------------------------------------------------------------
__global__ void __launch_bounds__(128)
gates_wmma_kernel(const float* __restrict__ x,
                  const float* __restrict__ gw,
                  const float* __restrict__ gb,
                  float* __restrict__ g) {
    __shared__ float xs[16 * XS_STR];

    const int b0  = blockIdx.x * 16;
    const int p   = blockIdx.y;
    const int tid = threadIdx.x;

#if HAVE_TDM
    // TDM: DMA the 16x256 x tile into LDS with 4-dword row padding (-> XS_STR).
    if (tid < 32) {
        tdm_load_tile(lds_byte_addr(xs), x + (size_t)b0 * INPUT_DIM,
                      /*dim0*/ INPUT_DIM, /*dim1*/ BATCH, /*dim2*/ 1,
                      /*tile0*/ INPUT_DIM, /*tile1*/ 16, /*tile2*/ 0,
                      /*stride0*/ INPUT_DIM, /*stride1*/ 0,
                      /*pad_interval*/ 7 /*256 dw*/, /*pad_amount*/ 3 /*4 dw*/, 1);
        __builtin_amdgcn_s_wait_tensorcnt(0);
    }
#else
    for (int j = tid * 4; j < 16 * INPUT_DIM; j += 128 * 4) {
        const int row = j >> 8;
        const int col = j & 255;
        *(float4*)&xs[row * XS_STR + col] =
            *(const float4*)&x[(size_t)(b0 + row) * INPUT_DIM + col];
    }
#endif
    __syncthreads();

    const int lane = tid & 31;
    const int hf   = lane >> 4;
    const int l16  = lane & 15;
    const int dcol = (tid >> 5) * 16 + l16;     // wave picks its 16-wide d tile

    const float* xrow  = &xs[l16 * XS_STR];
    const float* gwrow = &gw[((size_t)p * DIM + dcol) * INPUT_DIM];

    v8f acc = {};
    for (int kk = 0; kk < INPUT_DIM / 32; ++kk) {
        const int kb = kk * 32 + hf * 8;
        v16bf a, b;
        frag_from(a, xrow,  kb);   // A[m][k] = x[b0+m][k], m = l16 (ds b128)
        frag_from(b, gwrow, kb);   // B[k][n] = gate_w[p][dcol][k] (global b128)
        acc = __builtin_amdgcn_wmma_f32_16x16x32_bf16(
                  false, a, false, b, (short)0, acc, false, false);
    }

    const float bias = gb[p * DIM + dcol];
#pragma unroll
    for (int r = 0; r < 8; ++r) {
        const int m = r + hf * 8;  // C/D layout: M = r + half*8, N = lane%16
        g[((size_t)p * BATCH + (b0 + m)) * DIM + dcol] = FAST_TANH(acc[r] + bias);
    }
}

// ---------------------------------------------------------------------------
// Kernel 2: per-(pair,batch) 32x32 score tables via WMMA, then table lookup.
//   T[p,bi,ia,ib] = sum_k emb[rowA][ia][k] * g[p,b0+bi][k] * emb[rowB][ib][k]
//   out[b,n]      = sum_p T[p, bi, lab[colA_p], lab[colB_p]]
// grid BATCH/8, block 256 (8 waves). Dynamic LDS ~246 KB (WGP has 320 KB).
// ---------------------------------------------------------------------------
#define T_FLOATS   (48 * 1024)
#define EMB_FLOATS (N_HEADS * VOCAB * EMB_STR)
#define GT_FLOATS  (N_PAIR * 8 * DIM)
#define DYN_LDS_BYTES ((T_FLOATS + EMB_FLOATS + GT_FLOATS) * sizeof(float))

__global__ void __launch_bounds__(256)
score_table_kernel(const int* __restrict__ labels,
                   const float* __restrict__ emb,
                   const float* __restrict__ g,
                   float* __restrict__ out) {
    extern __shared__ float smem[];
    float* T    = smem;                 // [48][1024]
    float* embs = smem + T_FLOATS;      // [5*32][EMB_STR]
    float* gt   = embs + EMB_FLOATS;    // [6*8][64]

    const int b0  = blockIdx.x * 8;
    const int tid = threadIdx.x;

#if HAVE_TDM
    if (tid < 32) {
        // emb: 160 rows x 64 dwords, +4-dword pad per row (-> EMB_STR).
        tdm_load_tile(lds_byte_addr(embs), emb,
                      /*dim0*/ DIM, /*dim1*/ N_HEADS * VOCAB, /*dim2*/ 1,
                      /*tile0*/ DIM, /*tile1*/ N_HEADS * VOCAB, /*tile2*/ 0,
                      /*stride0*/ DIM, /*stride1*/ 0,
                      /*pad_interval*/ 5 /*64 dw*/, /*pad_amount*/ 3 /*4 dw*/, 1);
        // g tile: 3-D 64 x 8(bi) x 6(p); z stride = BATCH*DIM. Packed in LDS.
        tdm_load_tile(lds_byte_addr(gt), g + (size_t)b0 * DIM,
                      /*dim0*/ DIM, /*dim1*/ BATCH, /*dim2*/ N_PAIR,
                      /*tile0*/ DIM, /*tile1*/ 8, /*tile2*/ N_PAIR,
                      /*stride0*/ DIM, /*stride1*/ (unsigned long long)BATCH * DIM,
                      /*pad_interval*/ 0, /*pad_amount*/ 0, 0);
        __builtin_amdgcn_s_wait_tensorcnt(0);
    }
#else
    for (int j = tid * 4; j < N_HEADS * VOCAB * DIM; j += 256 * 4) {
        const int h   = j >> 11;
        const int rem = j & 2047;
        const int v   = rem >> 6;
        const int d   = rem & 63;
        *(float4*)&embs[(h * VOCAB + v) * EMB_STR + d] = *(const float4*)&emb[j];
    }
    for (int j = tid * 4; j < GT_FLOATS; j += 256 * 4) {
        const int p   = j >> 9;
        const int rem = j & 511;
        const int bi  = rem >> 6;
        const int d   = rem & 63;
        *(float4*)&gt[j] =
            *(const float4*)&g[((size_t)p * BATCH + (b0 + bi)) * DIM + d];
    }
#endif
    __syncthreads();

    const int wave = tid >> 5;
    const int lane = tid & 31;
    const int hf   = lane >> 4;
    const int l16  = lane & 15;

    // Phase 1: 48 tables (6 pairs x 8 batches), 6 per wave, 8 WMMAs each.
    for (int t = wave; t < N_PAIR * 8; t += 8) {
        const int p    = t >> 3;
        const int bi   = t & 7;
        const int rowA = (p < 3) ? 3 : 4;   // predictor : reorder
        const int rowB = p % 3;             // cf_perm / cf_primary / cf_secondary
        const float* ga = gt + ((p * 8 + bi) * DIM);
        float* Tt = T + t * 1024;

        // Hoist the gate vector (broadcast across lanes): 8 float4 LDS reads.
        float4 gv[2][4];
#pragma unroll
        for (int kk = 0; kk < 2; ++kk) {
            const int kb = kk * 32 + hf * 8;
            gv[kk][0] = *(const float4*)&ga[kb];
            gv[kk][1] = *(const float4*)&ga[kb + 4];
            gv[kk][2] = *(const float4*)&ga[kb + 16];
            gv[kk][3] = *(const float4*)&ga[kb + 20];
        }

        // Build all A fragments (gated emb rows) and B fragments once.
        v16bf afr[2][2], bfr[2][2];
#pragma unroll
        for (int mt = 0; mt < 2; ++mt) {
#pragma unroll
            for (int kk = 0; kk < 2; ++kk) {
                const int kb = kk * 32 + hf * 8;
                const float* er = &embs[(rowA * VOCAB + mt * 16 + l16) * EMB_STR];
                put4(afr[mt][kk], 0,  mul4(*(const float4*)&er[kb],      gv[kk][0]));
                put4(afr[mt][kk], 4,  mul4(*(const float4*)&er[kb + 4],  gv[kk][1]));
                put4(afr[mt][kk], 8,  mul4(*(const float4*)&er[kb + 16], gv[kk][2]));
                put4(afr[mt][kk], 12, mul4(*(const float4*)&er[kb + 20], gv[kk][3]));
            }
        }
#pragma unroll
        for (int nt = 0; nt < 2; ++nt) {
#pragma unroll
            for (int kk = 0; kk < 2; ++kk) {
                const int kb = kk * 32 + hf * 8;
                frag_from(bfr[nt][kk],
                          &embs[(rowB * VOCAB + nt * 16 + l16) * EMB_STR], kb);
            }
        }

        // 2x2 output tiles, K=64 in two WMMA steps each.
#pragma unroll
        for (int mt = 0; mt < 2; ++mt) {
#pragma unroll
            for (int nt = 0; nt < 2; ++nt) {
                v8f acc = {};
#pragma unroll
                for (int kk = 0; kk < 2; ++kk) {
                    acc = __builtin_amdgcn_wmma_f32_16x16x32_bf16(
                              false, afr[mt][kk], false, bfr[nt][kk],
                              (short)0, acc, false, false);
                }
#pragma unroll
                for (int r = 0; r < 8; ++r) {
                    const int ia = mt * 16 + r + hf * 8;
                    const int ib = nt * 16 + l16;
                    Tt[ia * VOCAB + ib] = acc[r];
                }
            }
        }
    }
    __syncthreads();

    // Phase 2: stream labels, 6 LDS gathers + adds per output.
    // Pair columns: colA in {0,0,0,4,4,4}, colB in {1,2,3,1,2,3}.
    for (int idx = tid; idx < 8 * NEGS; idx += 256) {
        const int bi = idx >> 9;
        const int n  = idx & (NEGS - 1);
        const int* lab = labels + ((size_t)(b0 + bi) * NEGS + n) * 6;

        const int next = idx + 256;
        if (next < 8 * NEGS) {
            const int* labn = labels +
                ((size_t)(b0 + (next >> 9)) * NEGS + (next & (NEGS - 1))) * 6;
            __builtin_prefetch(labn, 0, 1);   // global_prefetch_b8
        }

        const int l0 = lab[0] & 31;   // predictor
        const int l1 = lab[1] & 31;   // cf_perm
        const int l2 = lab[2] & 31;   // cf_primary
        const int l3 = lab[3] & 31;   // cf_secondary
        const int l4 = lab[4] & 31;   // reorder

        float s = 0.0f;
        s += T[((0 * 8 + bi) << 10) + (l0 << 5) + l1];
        s += T[((1 * 8 + bi) << 10) + (l0 << 5) + l2];
        s += T[((2 * 8 + bi) << 10) + (l0 << 5) + l3];
        s += T[((3 * 8 + bi) << 10) + (l4 << 5) + l1];
        s += T[((4 * 8 + bi) << 10) + (l4 << 5) + l2];
        s += T[((5 * 8 + bi) << 10) + (l4 << 5) + l3];

        out[(size_t)(b0 + bi) * NEGS + n] = s;
    }
}

// ---------------------------------------------------------------------------
// Host launch
// ---------------------------------------------------------------------------
extern "C" void kernel_launch(void* const* d_in, const int* in_sizes, int n_in,
                              void* d_out, int out_size, void* d_ws, size_t ws_size,
                              hipStream_t stream) {
    const float* x      = (const float*)d_in[0];
    const int*   labels = (const int*)  d_in[1];
    const float* emb    = (const float*)d_in[2];
    const float* gw     = (const float*)d_in[3];
    const float* gb     = (const float*)d_in[4];
    float* out = (float*)d_out;
    float* g   = (float*)d_ws;   // (6, 2048, 64) f32 = 3 MB

    (void)in_sizes; (void)n_in; (void)out_size; (void)ws_size;

    // Kernel 1: gates
    gates_wmma_kernel<<<dim3(BATCH / 16, N_PAIR), 128, 0, stream>>>(x, gw, gb, g);

    // Kernel 2: tables + lookup (needs ~246 KB dynamic LDS; WGP has 320 KB)
    (void)hipFuncSetAttribute((const void*)score_table_kernel,
                              hipFuncAttributeMaxDynamicSharedMemorySize,
                              (int)DYN_LDS_BYTES);
    score_table_kernel<<<BATCH / 8, 256, DYN_LDS_BYTES, stream>>>(labels, emb, g, out);
}
